// Decoder_8899172237774
// MI455X (gfx1250) — matrix operand order repaired and checked by reading, not compiled, not run on hardware
//
#include <hip/hip_runtime.h>
#include <hip/hip_bf16.h>
#include <math.h>

// ---------------------------------------------------------------- constants
constexpr int E   = 256;     // embedding dim
constexpr int H2  = 512;     // 2*E decoder hidden
constexpr int V   = 32000;   // vocab
constexpr int S   = 512;     // source length
constexpr int T   = 128;     // target length
constexpr int SOS = 1;

typedef __attribute__((ext_vector_type(16))) __bf16 v16bf;
typedef __attribute__((ext_vector_type(8)))  float  v8f;

__device__ __forceinline__ float sigmf(float x) { return 1.0f / (1.0f + __expf(-x)); }
__device__ __forceinline__ float reluf(float x) { return x > 0.0f ? x : 0.0f; }

// =================================================================
// WMMA GEMM, register-level B reuse + fused B-type conversion:
//   C[M,N] = A[M,K](bf16,row-major) * B (+bias1)(+bias2)
//   BT=true : B is a weight W[N,K] row-major -> computes A @ W^T
//   BT=false: B is [K,N] row-major           -> computes A @ B
//   TB = __bf16 or float; float B converts in-register (v_cvt_pk_bf16_f32),
//   so f32 weights stream from HBM in a SINGLE pass (131 MB once for lin_W
//   ~5.6 us at 23.3 TB/s, vs 261 MB for convert-then-GEMM).
// Each wave owns MT vertically-stacked 16x16 tiles of one N-column strip:
// every B fragment is loaded ONCE and multiplied into MT register-resident
// accumulators. The empty memory-clobber barrier keeps the fragment loads
// grouped into clauses ahead of the WMMAs; stronger "+v" operand-tying
// fences were tried (Rounds 5-6) and consistently caused accumulator
// spills to scratch in the hot loop, so this lighter form is final: the
// compiler's schedule (1-deep load/WMMA software pipeline, rotating address
// registers, no spills, no hazard NOPs) plus ~8 waves/SIMD occupancy covers
// the residual load latency of this HBM-bound streaming GEMM.
// Fragment packing: A and B both use contiguous k = 16*half + s (valid since they match).
// Requires M % (16*MT) == 0, N % 16 == 0, K % 32 == 0.
// =================================================================
template <bool BT, int MT, typename TB>
__global__ void wmma_gemm_bf16(const __bf16* __restrict__ A,
                               const TB* __restrict__ B,
                               const float* __restrict__ bias1,
                               const float* __restrict__ bias2,
                               float* __restrict__ C,
                               __bf16* __restrict__ Cbf,  // optional bf16 copy
                               int M, int N, int K) {
  const int wavesPerBlk = blockDim.x >> 5;
  const int wave    = blockIdx.x * wavesPerBlk + ((int)threadIdx.x >> 5);
  const int tilesN  = N >> 4;
  const int mGroups = M / (16 * MT);
  if (wave >= mGroups * tilesN) return;  // uniform per wave -> EXEC all-ones

  const int tn     = (wave % tilesN) << 4;
  const int tmBase = (wave / tilesN) * (16 * MT);

  const int lane = threadIdx.x & 31;
  const int half = lane >> 4;   // 0: lanes 0-15, 1: lanes 16-31
  const int l    = lane & 15;

  v8f acc[MT];
#pragma unroll
  for (int i = 0; i < MT; ++i) acc[i] = (v8f){};

  for (int k0 = 0; k0 < K; k0 += 32) {
    // ---- B fragment 32x16 (loaded once per k-step, reused MT times);
    //      TB=float converts in-register (v_cvt co-executes with WMMA) ----
    v16bf b;
    {
      const int n  = tn + l;
      const int kb = k0 + half * 16;
      if (BT) {
        const TB* bp = B + (size_t)n * K + kb;
        __builtin_prefetch((const void*)(bp + 2048), 0, 0);  // global_prefetch_b8
#pragma unroll
        for (int j = 0; j < 8; ++j) {
          b[2 * j]     = (__bf16)bp[2 * j];
          b[2 * j + 1] = (__bf16)bp[2 * j + 1];
        }
      } else {
#pragma unroll
        for (int j = 0; j < 8; ++j) {
          b[2 * j]     = (__bf16)B[(size_t)(kb + 2 * j) * N + n];
          b[2 * j + 1] = (__bf16)B[(size_t)(kb + 2 * j + 1) * N + n];
        }
      }
    }
    // ---- preload MT A fragments (A small & L2-resident) ----
    v16bf a[MT];
#pragma unroll
    for (int i = 0; i < MT; ++i) {
      const __bf16* ap = A + (size_t)(tmBase + i * 16 + l) * K + k0 + half * 16;
#pragma unroll
      for (int j = 0; j < 4; ++j) {          // VGPR 0..3 -> K = base+0..7
        a[i][2 * j]     = ap[2 * j];
        a[i][2 * j + 1] = ap[2 * j + 1];
      }
#pragma unroll
      for (int j = 0; j < 4; ++j) {          // VGPR 4..7 -> K = base+8..15
        a[i][8 + 2 * j] = ap[8 + 2 * j];
        a[i][9 + 2 * j] = ap[8 + 2 * j + 1];
      }
    }
    // Light scheduling fence: keeps fragment loads clause-grouped ahead of
    // the WMMA chain without constraining register allocation (stronger
    // operand-tying fences caused scratch spills of the accumulators).
    asm volatile("" ::: "memory");
    // ---- MT back-to-back WMMAs sharing one B fragment ----
#pragma unroll
    for (int i = 0; i < MT; ++i)
      acc[i] = __builtin_amdgcn_wmma_f32_16x16x32_bf16(false, a[i], false, b,
                                                       (short)0, acc[i], false, false);
  }

  // ---- store: tile i, VGPR r -> m = tmBase + i*16 + r + half*8, n = tn+l ----
  const int n = tn + l;
  float bsum = 0.0f;
  if (bias1) bsum += bias1[n];
  if (bias2) bsum += bias2[n];
#pragma unroll
  for (int i = 0; i < MT; ++i) {
#pragma unroll
    for (int r = 0; r < 8; ++r) {
      const int m = tmBase + i * 16 + r + half * 8;
      const float v = acc[i][r] + bsum;
      C[(size_t)m * N + n] = v;
      if (Cbf) Cbf[(size_t)m * N + n] = (__bf16)v;
    }
  }
}

// ============================ elementwise / prep =========================

// X_bf[s][e] = bf16(relu(enc_emb[tokens[s]][e]))
__global__ void enc_embed_k(const int* __restrict__ tokens,
                            const float* __restrict__ emb,
                            __bf16* __restrict__ Xbf) {
  int s = blockIdx.x, e = threadIdx.x;  // 512 blocks x 256 threads
  float v = emb[(size_t)tokens[s] * E + e];
  Xbf[(size_t)s * E + e] = (__bf16)reluf(v);
}

// ==================== encoder bidirectional LSTM scan ====================
// gridDim.x = 2 (dir 0 = fwd, dir 1 = bwd), 256 threads; gate order i,f,g,o.
// gates_in already holds  x@Wih^T + bih + bhh  for every source position.
__global__ void enc_lstm_scan_k(const float* __restrict__ gates_f,
                                const float* __restrict__ gates_b,
                                const float* __restrict__ Whh_f,
                                const float* __restrict__ Whh_b,
                                float* __restrict__ encoded,     // [S][2E]
                                float* __restrict__ enc_hidden) { // [2E]
  const int dir = blockIdx.x;
  const float* G   = dir ? gates_b : gates_f;
  const float* Whh = dir ? Whh_b : Whh_f;
  __shared__ float h[E];
  const int j = threadIdx.x;
  h[j] = 0.0f;
  float c = 0.0f;
  __syncthreads();
  for (int t = 0; t < S; ++t) {
    const int row = dir ? (S - 1 - t) : t;           // source position
    const float* g = G + (size_t)row * (4 * E);
    float gi = g[j], gf = g[E + j], gg = g[2 * E + j], go = g[3 * E + j];
    const float* wi = Whh + (size_t)(0 * E + j) * E;
    const float* wf = Whh + (size_t)(1 * E + j) * E;
    const float* wg = Whh + (size_t)(2 * E + j) * E;
    const float* wo = Whh + (size_t)(3 * E + j) * E;
#pragma unroll 4
    for (int k = 0; k < E; ++k) {
      const float hk = h[k];
      gi += wi[k] * hk; gf += wf[k] * hk; gg += wg[k] * hk; go += wo[k] * hk;
    }
    c = sigmf(gf) * c + sigmf(gi) * tanhf(gg);
    const float hn = sigmf(go) * tanhf(c);
    __syncthreads();
    h[j] = hn;
    encoded[(size_t)row * (2 * E) + dir * E + j] = hn;  // bwd writes reversed back
    __syncthreads();
  }
  enc_hidden[dir * E + j] = h[j];  // final h of each direction
}

// enc_m_bf[s][j] = mask(s) ? bf16(encoded[s][j]) : 0
__global__ void mask_enc_k(const float* __restrict__ encoded,
                           const int* __restrict__ tokens,
                           __bf16* __restrict__ enc_m_bf) {
  int s = blockIdx.x, j = threadIdx.x;  // 512 x 512
  float v = (tokens[s] > 0) ? encoded[(size_t)s * H2 + j] : 0.0f;
  enc_m_bf[(size_t)s * H2 + j] = (__bf16)v;
}

// decoder inputs for all steps (teacher forcing -> known in advance)
__global__ void dec_xin_k(const int* __restrict__ trg,
                          const float* __restrict__ dec_emb,
                          const float* __restrict__ enc_hidden,
                          __bf16* __restrict__ xin,   // [T][4E]
                          float* __restrict__ e2f) {  // [T][2E]
  int t = blockIdx.x, j = threadIdx.x;  // 128 x 512
  int tok = (t == 0) ? SOS : trg[t - 1];
  float v = reluf(dec_emb[(size_t)tok * H2 + j]);
  e2f[(size_t)t * H2 + j] = v;
  xin[(size_t)t * (2 * H2) + j]      = (__bf16)v;
  xin[(size_t)t * (2 * H2) + H2 + j] = (__bf16)enc_hidden[j];
}

// ====================== decoder LSTM scan (sequential) ===================
// one block, 512 threads. gates_pre = xin@dWih^T + dbih + dbhh precomputed.
__global__ void dec_lstm_scan_k(const float* __restrict__ gates_pre, // [T][4*H2]
                                const float* __restrict__ dWhh,      // [4*H2][H2]
                                const float* __restrict__ enc_hidden,
                                float* __restrict__ H,               // [T][H2]
                                __bf16* __restrict__ Hbf) {
  __shared__ float h[H2];
  const int j = threadIdx.x;
  h[j] = enc_hidden[j];  // decoder h0 = encoder final hidden, c0 = 0
  float c = 0.0f;
  __syncthreads();
  for (int t = 0; t < T; ++t) {
    const float* g = gates_pre + (size_t)t * (4 * H2);
    float gi = g[j], gf = g[H2 + j], gg = g[2 * H2 + j], go = g[3 * H2 + j];
    const float* wi = dWhh + (size_t)(0 * H2 + j) * H2;
    const float* wf = dWhh + (size_t)(1 * H2 + j) * H2;
    const float* wg = dWhh + (size_t)(2 * H2 + j) * H2;
    const float* wo = dWhh + (size_t)(3 * H2 + j) * H2;
#pragma unroll 4
    for (int k = 0; k < H2; ++k) {
      const float hk = h[k];
      gi += wi[k] * hk; gf += wf[k] * hk; gg += wg[k] * hk; go += wo[k] * hk;
    }
    c = sigmf(gf) * c + sigmf(gi) * tanhf(gg);
    const float hn = sigmf(go) * tanhf(c);
    __syncthreads();
    h[j] = hn;
    H[(size_t)t * H2 + j]   = hn;
    Hbf[(size_t)t * H2 + j] = (__bf16)hn;
    __syncthreads();
  }
}

// ============== masked attention softmax over S (per step t) =============
__global__ void attn_softmax_k(const float* __restrict__ scores, // [T][S]
                               const int* __restrict__ tokens,
                               float* __restrict__ aw,           // [T][S]
                               __bf16* __restrict__ awbf) {
  const int t = blockIdx.x, s = threadIdx.x;  // 128 x 512
  __shared__ float red[512];
  const bool valid = tokens[s] > 0;
  const float sc = valid ? scores[(size_t)t * S + s] : -INFINITY;
  red[s] = sc; __syncthreads();
  for (int w = 256; w > 0; w >>= 1) {
    if (s < w) red[s] = fmaxf(red[s], red[s + w]);
    __syncthreads();
  }
  const float mx = red[0]; __syncthreads();
  const float e = (valid && mx > -INFINITY) ? __expf(sc - mx) : 0.0f;
  red[s] = e; __syncthreads();
  for (int w = 256; w > 0; w >>= 1) {
    if (s < w) red[s] += red[s + w];
    __syncthreads();
  }
  const float sum = red[0];
  const float a = (sum > 0.0f) ? e / sum : 0.0f;  // NaN-guard of reference
  aw[(size_t)t * S + s]   = a;
  awbf[(size_t)t * S + s] = (__bf16)a;
}

// hc_bf[t] = concat(Hbf[t], ctxbf[t])
__global__ void concat_hc_k(const __bf16* __restrict__ Hbf,
                            const __bf16* __restrict__ ctxbf,
                            __bf16* __restrict__ hcbf) {
  int t = blockIdx.x, j = threadIdx.x;  // 128 x 512
  hcbf[(size_t)t * (2 * H2) + j]      = Hbf[(size_t)t * H2 + j];
  hcbf[(size_t)t * (2 * H2) + H2 + j] = ctxbf[(size_t)t * H2 + j];
}

// p_gen[t] = sigmoid(copy_W . [h, context, e2] + copy_b)
__global__ void pgen_k(const float* __restrict__ H, const float* __restrict__ ctx,
                       const float* __restrict__ e2, const float* __restrict__ copy_W,
                       const float* __restrict__ copy_b, float* __restrict__ pgen) {
  const int t = blockIdx.x, j = threadIdx.x;  // 128 x 512
  __shared__ float red[512];
  float p = copy_W[j]            * H[(size_t)t * H2 + j]
          + copy_W[H2 + j]       * ctx[(size_t)t * H2 + j]
          + copy_W[2 * H2 + j]   * e2[(size_t)t * H2 + j];
  red[j] = p; __syncthreads();
  for (int w = 256; w > 0; w >>= 1) {
    if (j < w) red[j] += red[j + w];
    __syncthreads();
  }
  if (j == 0) pgen[t] = sigmf(red[0] + copy_b[0]);
}

// softmax over vocab, then  out[t][v] = pgen[t] * softmax(logits[t])[v]
__global__ void vocab_softmax_k(const float* __restrict__ logits,  // [T][V]
                                const float* __restrict__ pgen,
                                float* __restrict__ out) {         // [T][V]
  const int t = blockIdx.x, tid = threadIdx.x;  // 128 x 256
  __shared__ float red[256];
  const float* row = logits + (size_t)t * V;
  float mx = -INFINITY;
  for (int v = tid; v < V; v += 256) mx = fmaxf(mx, row[v]);
  red[tid] = mx; __syncthreads();
  for (int w = 128; w > 0; w >>= 1) {
    if (tid < w) red[tid] = fmaxf(red[tid], red[tid + w]);
    __syncthreads();
  }
  mx = red[0]; __syncthreads();
  float sum = 0.0f;
  for (int v = tid; v < V; v += 256) sum += __expf(row[v] - mx);
  red[tid] = sum; __syncthreads();
  for (int w = 128; w > 0; w >>= 1) {
    if (tid < w) red[tid] += red[tid + w];
    __syncthreads();
  }
  const float inv = pgen[t] / red[0];
  for (int v = tid; v < V; v += 256)
    out[(size_t)t * V + v] = __expf(row[v] - mx) * inv;
}

// out[t][tokens[s]] += (1-pgen[t]) * aw[t][s]   (copy distribution scatter)
__global__ void copy_scatter_k(const int* __restrict__ tokens,
                               const float* __restrict__ aw,
                               const float* __restrict__ pgen,
                               float* __restrict__ out) {
  const int t = blockIdx.x, s = threadIdx.x;  // 128 x 512
  const float v = (1.0f - pgen[t]) * aw[(size_t)t * S + s];
  atomicAdd(&out[(size_t)t * V + tokens[s]], v);
}

// log_prob = sum_t log(out[t][trg[t]])
__global__ void logprob_k(const float* __restrict__ out, const int* __restrict__ trg,
                          float* __restrict__ dst) {
  const int t = threadIdx.x;  // 1 block x 128
  __shared__ float red[T];
  red[t] = logf(out[(size_t)t * V + trg[t]]);
  __syncthreads();
  for (int w = 64; w > 0; w >>= 1) {
    if (t < w) red[t] += red[t + w];
    __syncthreads();
  }
  if (t == 0) dst[0] = red[0];
}

// ================================ launcher ===============================
constexpr int MT = 8;  // M-tiles per wave (128 rows per wave strip)
static inline int gemm_blocks(int M, int N) {
  return ((M / (16 * MT)) * (N / 16) + 3) / 4;
}

extern "C" void kernel_launch(void* const* d_in, const int* in_sizes, int n_in,
                              void* d_out, int out_size, void* d_ws, size_t ws_size,
                              hipStream_t stream) {
  const int*   tokens  = (const int*)d_in[0];
  const int*   trg     = (const int*)d_in[1];
  const float* enc_emb = (const float*)d_in[2];
  const float* dec_emb = (const float*)d_in[3];
  const float* Wih_f = (const float*)d_in[4],  *Whh_f = (const float*)d_in[5];
  const float* bih_f = (const float*)d_in[6],  *bhh_f = (const float*)d_in[7];
  const float* Wih_b = (const float*)d_in[8],  *Whh_b = (const float*)d_in[9];
  const float* bih_b = (const float*)d_in[10], *bhh_b = (const float*)d_in[11];
  const float* dWih  = (const float*)d_in[12], *dWhh  = (const float*)d_in[13];
  const float* dbih  = (const float*)d_in[14], *dbhh  = (const float*)d_in[15];
  const float* attn_W = (const float*)d_in[16], *attn_b = (const float*)d_in[17];
  const float* W_W    = (const float*)d_in[18], *W_b    = (const float*)d_in[19];
  const float* lin_W  = (const float*)d_in[20], *lin_b  = (const float*)d_in[21];
  const float* copy_W = (const float*)d_in[22], *copy_b = (const float*)d_in[23];
  float* out = (float*)d_out;  // [T*V] p_all, then [1] log_prob

  // ---- workspace carve-up (~30 MB; deterministic, rewritten every call) ----
  char* p = (char*)d_ws;
  auto alloc = [&](size_t bytes) {
    void* r = (void*)p;
    p += (bytes + 255) & ~(size_t)255;
    return r;
  };
  float*  logits   = (float*) alloc((size_t)T * V * 4);           // 16.4 MB
  __bf16* Xbf      = (__bf16*)alloc((size_t)S * E * 2);
  float*  gates_f  = (float*) alloc((size_t)S * 4 * E * 4);
  float*  gates_b  = (float*) alloc((size_t)S * 4 * E * 4);
  float*  encoded  = (float*) alloc((size_t)S * H2 * 4);
  float*  enc_hid  = (float*) alloc((size_t)H2 * 4);
  __bf16* encm_bf  = (__bf16*)alloc((size_t)S * H2 * 2);
  __bf16* xin_bf   = (__bf16*)alloc((size_t)T * 4 * E * 2);
  float*  e2f      = (float*) alloc((size_t)T * H2 * 4);
  float*  gatespre = (float*) alloc((size_t)T * 4 * H2 * 4);
  float*  Hf       = (float*) alloc((size_t)T * H2 * 4);
  __bf16* Hbf      = (__bf16*)alloc((size_t)T * H2 * 2);
  float*  Qf       = (float*) alloc((size_t)T * H2 * 4);
  __bf16* Qbf      = (__bf16*)alloc((size_t)T * H2 * 2);
  float*  scores   = (float*) alloc((size_t)T * S * 4);
  float*  aw       = (float*) alloc((size_t)T * S * 4);
  __bf16* awbf     = (__bf16*)alloc((size_t)T * S * 2);
  float*  ctxf     = (float*) alloc((size_t)T * H2 * 4);
  __bf16* ctxbf    = (__bf16*)alloc((size_t)T * H2 * 2);
  __bf16* hcbf     = (__bf16*)alloc((size_t)T * 2 * H2 * 2);
  float*  outhc    = (float*) alloc((size_t)T * 2 * H2 * 4);
  __bf16* outhcbf  = (__bf16*)alloc((size_t)T * 2 * H2 * 2);
  float*  pgen     = (float*) alloc((size_t)T * 4);
  (void)ws_size; (void)n_in; (void)in_sizes; (void)out_size;

  // ---- encoder: batched input projection + sequential scan ----
  enc_embed_k<<<S, E, 0, stream>>>(tokens, enc_emb, Xbf);
  wmma_gemm_bf16<true, MT, float><<<gemm_blocks(S, 4 * E), 128, 0, stream>>>(
      Xbf, Wih_f, bih_f, bhh_f, gates_f, nullptr, S, 4 * E, E);
  wmma_gemm_bf16<true, MT, float><<<gemm_blocks(S, 4 * E), 128, 0, stream>>>(
      Xbf, Wih_b, bih_b, bhh_b, gates_b, nullptr, S, 4 * E, E);
  enc_lstm_scan_k<<<2, E, 0, stream>>>(gates_f, gates_b, Whh_f, Whh_b,
                                       encoded, enc_hid);
  mask_enc_k<<<S, H2, 0, stream>>>(encoded, tokens, encm_bf);

  // ---- decoder: batched input projection + sequential scan ----
  dec_xin_k<<<T, H2, 0, stream>>>(trg, dec_emb, enc_hid, xin_bf, e2f);
  wmma_gemm_bf16<true, MT, float><<<gemm_blocks(T, 4 * H2), 128, 0, stream>>>(
      xin_bf, dWih, dbih, dbhh, gatespre, nullptr, T, 4 * H2, 4 * E);
  dec_lstm_scan_k<<<1, H2, 0, stream>>>(gatespre, dWhh, enc_hid, Hf, Hbf);

  // ---- batched attention ----
  // Q = H @ attn_W^T + attn_b                 [T,H2]
  wmma_gemm_bf16<true, MT, float><<<gemm_blocks(T, H2), 128, 0, stream>>>(
      Hbf, attn_W, attn_b, nullptr, Qf, Qbf, T, H2, H2);
  // scores = Q @ enc_m^T                      [T,S]  (enc_m as W[N=S,K=H2])
  wmma_gemm_bf16<true, MT, __bf16><<<gemm_blocks(T, S), 128, 0, stream>>>(
      Qbf, encm_bf, nullptr, nullptr, scores, nullptr, T, S, H2);
  attn_softmax_k<<<T, S, 0, stream>>>(scores, tokens, aw, awbf);
  // context = aw @ enc_m                      [T,H2] (enc_m as B[K=S,N=H2])
  wmma_gemm_bf16<false, MT, __bf16><<<gemm_blocks(T, H2), 128, 0, stream>>>(
      awbf, encm_bf, nullptr, nullptr, ctxf, ctxbf, T, H2, S);

  // ---- output head ----
  concat_hc_k<<<T, H2, 0, stream>>>(Hbf, ctxbf, hcbf);
  wmma_gemm_bf16<true, MT, float><<<gemm_blocks(T, 2 * H2), 128, 0, stream>>>(
      hcbf, W_W, W_b, nullptr, outhc, outhcbf, T, 2 * H2, 2 * H2);
  // vocab projection: 8.4 GFLOP, single 131 MB pass over f32 lin_W   [T,V]
  wmma_gemm_bf16<true, MT, float><<<gemm_blocks(T, V), 128, 0, stream>>>(
      outhcbf, lin_W, lin_b, nullptr, logits, nullptr, T, V, 2 * H2);

  // ---- copy mechanism + final mixture ----
  pgen_k<<<T, H2, 0, stream>>>(Hf, ctxf, e2f, copy_W, copy_b, pgen);
  vocab_softmax_k<<<T, 256, 0, stream>>>(logits, pgen, out);  // out = pgen*p_word
  copy_scatter_k<<<T, S, 0, stream>>>(tokens, aw, pgen, out); // += (1-pgen)*p_copy
  logprob_k<<<1, T, 0, stream>>>(out, trg, out + (size_t)T * V);
}
